// WaveBlock_67671504716380
// MI455X (gfx1250) — matrix operand order repaired and checked by reading, not compiled, not run on hardware
//
#include <hip/hip_runtime.h>
#include <math.h>

// ---------------------------------------------------------------------------
// WaveBlock fused pipeline for MI455X (gfx1250, wave32, WMMA).
// Big GEMMs (x@att and y@ffW^T) use v_wmma_f32_16x16x32_f16 with f32 accum.
// f16 chosen over bf16: same WMMA shape/throughput, better precision (10-bit
// mantissa), and single-instruction packed convert (v_cvt_pkrtz_f16_f32).
// ---------------------------------------------------------------------------

typedef __attribute__((ext_vector_type(16))) _Float16      v16h;
typedef __attribute__((ext_vector_type(8)))  float         v8f;
typedef __attribute__((ext_vector_type(8)))  unsigned int  v8u;

constexpr int BB   = 32;
constexpr int LL   = 4096;
constexpr int FF   = 128;
constexpr int NLEV = 12;
constexpr int CHROWS = 256;            // wavelet L-chunk
constexpr int NCHQ   = LL / CHROWS;    // 16
constexpr int NCHL   = LL / 128;       // 32 GEMM row-chunks per batch
#define EPSV 1e-5f

__device__ __forceinline__ unsigned short f2h(float f) {
  return __builtin_bit_cast(unsigned short, (_Float16)f);
}

// Pack two f32 into one dword of f16 (single v_cvt_pkrtz_f16_f32).
__device__ __forceinline__ unsigned int pk2h(float a, float b) {
#if __has_builtin(__builtin_amdgcn_cvt_pkrtz)
  return __builtin_bit_cast(unsigned int, __builtin_amdgcn_cvt_pkrtz(a, b));
#else
  return (unsigned int)f2h(a) | ((unsigned int)f2h(b) << 16);
#endif
}

// --------------------------- small prep kernels ----------------------------

__global__ void prep_ffw(const float* __restrict__ w, unsigned short* __restrict__ wb) {
  int i = blockIdx.x * blockDim.x + threadIdx.x;
  if (i < FF * FF) wb[i] = f2h(w[i]);
}

// Wavelet partial reduction: q[b,f] = sum_l c_l * x[b,l,f]  (chunked over L)
__global__ void wave_partial(const float* __restrict__ x,
                             const float* __restrict__ qw, const float* __restrict__ kw,
                             float* __restrict__ qpart, float* __restrict__ kpart) {
  __shared__ float scq[CHROWS], sck[CHROWS];
  const int ch = blockIdx.x, b = blockIdx.y, f = threadIdx.x;
  for (int r = f; r < CHROWS; r += blockDim.x) {
    int l = ch * CHROWS + r;
    float cq = 1.f, ck = 1.f;
#pragma unroll
    for (int i = 0; i < NLEV; i++) {
      int bit = (l >> i) & 1;
      cq *= qw[i * 2 + bit];
      ck *= kw[i * 2 + bit];
    }
    scq[r] = cq; sck[r] = ck;
  }
  __syncthreads();
  const float* xb = x + ((size_t)b * LL + (size_t)ch * CHROWS) * FF + f;
  float aq = 0.f, ak = 0.f;
  for (int r = 0; r < CHROWS; r++) {
    float xv = xb[(size_t)r * FF];
    aq = fmaf(scq[r], xv, aq);
    ak = fmaf(sck[r], xv, ak);
  }
  qpart[((size_t)b * NCHQ + ch) * FF + f] = aq;
  kpart[((size_t)b * NCHQ + ch) * FF + f] = ak;
}

__global__ void wave_combine(const float* __restrict__ qpart, const float* __restrict__ kpart,
                             const float* __restrict__ qw, const float* __restrict__ qb,
                             const float* __restrict__ kw, const float* __restrict__ kbv,
                             float* __restrict__ qv, float* __restrict__ kv) {
  int b = blockIdx.x, f = threadIdx.x;
  float q = 0.f, k = 0.f;
  for (int c = 0; c < NCHQ; c++) {
    q += qpart[((size_t)b * NCHQ + c) * FF + f];
    k += kpart[((size_t)b * NCHQ + c) * FF + f];
  }
  float cq = 0.f, ck = 0.f;
#pragma unroll
  for (int i = 0; i < NLEV; i++) {
    cq = (qw[2 * i] + qw[2 * i + 1]) * cq + qb[i];
    ck = (kw[2 * i] + kw[2 * i + 1]) * ck + kbv[i];
  }
  qv[b * FF + f] = q + cq;
  kv[b * FF + f] = k + ck;
}

// att[f,g] = softmax_g(q_f * k_g / sqrt(F)); store TRANSPOSED in f16:
// attT[b][g][f] so the WMMA B-fragment (k=f contiguous per lane n=g) is a
// pair of 16-byte loads per lane, 32 bytes apart.
__global__ void att_softmax(const float* __restrict__ qv, const float* __restrict__ kv,
                            unsigned short* __restrict__ attT) {
  __shared__ float sk[FF];
  int b = blockIdx.x, f = threadIdx.x;
  sk[f] = kv[b * FF + f];
  __syncthreads();
  const float scale = 0.08838834764831845f;   // 1/sqrt(128)
  float qf = qv[b * FF + f] * scale;
  float mx = -3.4e38f;
  for (int g = 0; g < FF; g++) mx = fmaxf(mx, qf * sk[g]);
  float sum = 0.f;
  for (int g = 0; g < FF; g++) sum += expf(qf * sk[g] - mx);
  float inv = 1.f / sum;
  for (int g = 0; g < FF; g++)
    attT[((size_t)b * FF + g) * FF + f] = f2h(expf(qf * sk[g] - mx) * inv);
}

// --------------------------- fused WMMA GEMM -------------------------------
// MODE 0: Y = X + X @ att            (B-matrix = attT[b], residual = X)
// MODE 1: Z = Yn + Yn @ W^T + ffb    (Yn = Y*a + c applied on the fly; in-place)
// Also emits deterministic per-block (sum, sumsq) column partials for the
// following instance-norm. 256 thr = 8 waves, each wave owns 16 rows x 128 cols.
template <int MODE>
__global__ void __launch_bounds__(256)
gemm_fused(const float* __restrict__ X, const unsigned short* __restrict__ Bm,
           const float* __restrict__ nA, const float* __restrict__ nC,
           const float* __restrict__ ffb, float* __restrict__ Y,
           float* __restrict__ statp) {
  __shared__ float sA[FF], sC[FF];
  __shared__ float sStat[16][FF][2];          // [wave*2+hi][col][sum,sumsq]
  const int rc = blockIdx.x, b = blockIdx.y;
  const int tid = threadIdx.x, wave = tid >> 5, lane = tid & 31;
  const int m = lane & 15, hi = lane >> 4;
  const int rowbase = rc * 128 + wave * 16;
  const float* Xb = X + ((size_t)b * LL + rowbase) * FF;
  float*       Yb = Y + ((size_t)b * LL + rowbase) * FF;
  const unsigned short* Bb = Bm + (MODE == 0 ? (size_t)b * FF * FF : 0);

  if constexpr (MODE == 1) {
    if (tid < FF) { sA[tid] = nA[b * FF + tid]; sC[tid] = nC[b * FF + tid]; }
  }
  __syncthreads();

  v8f acc[8];
#pragma unroll
  for (int j = 0; j < 8; j++)
#pragma unroll
    for (int e = 0; e < 8; e++) acc[j][e] = 0.f;

  const float* xrow = Xb + (size_t)m * FF;    // A rows: lane m (both halves)
#pragma unroll
  for (int kb = 0; kb < 4; kb++) {
    // A fragment 16x32 f16 (ISA layout): lane half `hi` holds
    // k_local in [hi*8, hi*8+8) and [16+hi*8, 16+hi*8+8)
    const int k0 = kb * 32 + hi * 8;
    float4 a0 = *(const float4*)(xrow + k0);
    float4 a1 = *(const float4*)(xrow + k0 + 4);
    float4 b0 = *(const float4*)(xrow + k0 + 16);
    float4 b1 = *(const float4*)(xrow + k0 + 20);
    float c0[8] = {a0.x, a0.y, a0.z, a0.w, a1.x, a1.y, a1.z, a1.w};
    float c1[8] = {b0.x, b0.y, b0.z, b0.w, b1.x, b1.y, b1.z, b1.w};
    if constexpr (MODE == 1) {
#pragma unroll
      for (int j = 0; j < 8; j++) {
        c0[j] = c0[j] * sA[k0 + j]      + sC[k0 + j];
        c1[j] = c1[j] * sA[k0 + 16 + j] + sC[k0 + 16 + j];
      }
    }
    v8u au;
#pragma unroll
    for (int j = 0; j < 4; j++) {
      au[j]     = pk2h(c0[2 * j], c0[2 * j + 1]);
      au[4 + j] = pk2h(c1[2 * j], c1[2 * j + 1]);
    }
    v16h af = __builtin_bit_cast(v16h, au);

#pragma unroll
    for (int jt = 0; jt < 8; jt++) {
      // B fragment 32x16 f16: lane n=m, half hi holds k_local in [hi*8, hi*8+8) and [16+hi*8, 16+hi*8+8)
      const unsigned short* brow = Bb + (size_t)(jt * 16 + m) * FF + kb * 32 + hi * 8;
      const uint4* bp = (const uint4*)brow;
      uint4 p0 = bp[0], p1 = bp[2];
      v8u bu;
      bu[0] = p0.x; bu[1] = p0.y; bu[2] = p0.z; bu[3] = p0.w;
      bu[4] = p1.x; bu[5] = p1.y; bu[6] = p1.z; bu[7] = p1.w;
      v16h bf = __builtin_bit_cast(v16h, bu);
      acc[jt] = __builtin_amdgcn_wmma_f32_16x16x32_f16(
          false, af, false, bf, (short)0, acc[jt], false, false);
    }
  }

  // Epilogue: residual (+bias/norm), store, per-column partial stats.
  // D layout: lane = n + 16*(m>=8); VGPR v -> row m = hi*8+v, col = jt*16+n.
#pragma unroll
  for (int jt = 0; jt < 8; jt++) {
    const int col = jt * 16 + m;
    float s = 0.f, ss = 0.f;
#pragma unroll
    for (int v = 0; v < 8; v++) {
      const int r = hi * 8 + v;
      float res;
      if constexpr (MODE == 0) {
        res = Xb[(size_t)r * FF + col];
      } else {
        float yv = Xb[(size_t)r * FF + col];
        res = yv * sA[col] + sC[col] + ffb[col];
      }
      float o = acc[jt][v] + res;
      Yb[(size_t)r * FF + col] = o;
      s += o; ss += o * o;
    }
    sStat[wave * 2 + hi][col][0] = s;
    sStat[wave * 2 + hi][col][1] = ss;
  }
  __syncthreads();
  {
    const int col = tid >> 1, st = tid & 1;
    float t = 0.f;
#pragma unroll
    for (int sl = 0; sl < 16; sl++) t += sStat[sl][col][st];
    statp[(((size_t)b * NCHL + rc) * FF + col) * 2 + st] = t;
  }
}

// Combine per-chunk stats -> per-(b,f) scale/shift: yn = y*a + c
__global__ void stats_combine(const float* __restrict__ statp,
                              const float* __restrict__ gamma, const float* __restrict__ beta,
                              float* __restrict__ nA, float* __restrict__ nC) {
  int b = blockIdx.x, f = threadIdx.x;
  double s = 0.0, ss = 0.0;
  for (int c = 0; c < NCHL; c++) {
    s  += (double)statp[(((size_t)b * NCHL + c) * FF + f) * 2 + 0];
    ss += (double)statp[(((size_t)b * NCHL + c) * FF + f) * 2 + 1];
  }
  double mean = s / (double)LL;
  double var  = ss / (double)LL - mean * mean;
  float a = gamma[f] * (float)(1.0 / sqrt(var + (double)EPSV));
  nA[b * FF + f] = a;
  nC[b * FF + f] = beta[f] - (float)mean * a;
}

// Final in-place elementwise normalize of d_out
__global__ void final_norm(float* __restrict__ out,
                           const float* __restrict__ nA, const float* __restrict__ nC) {
  size_t i4 = (size_t)blockIdx.x * blockDim.x + threadIdx.x;
  size_t total4 = (size_t)BB * LL * FF / 4;
  if (i4 >= total4) return;
  float4* p = (float4*)out + i4;
  size_t base = i4 * 4;
  int f = (int)(base & (FF - 1));
  int b = (int)(base >> 19);                 // L*F = 2^19
  const float* a = nA + b * FF + f;
  const float* c = nC + b * FF + f;
  float4 v = *p;
  v.x = v.x * a[0] + c[0];
  v.y = v.y * a[1] + c[1];
  v.z = v.z * a[2] + c[2];
  v.w = v.w * a[3] + c[3];
  *p = v;
}

// ------------------------------ launcher -----------------------------------

extern "C" void kernel_launch(void* const* d_in, const int* in_sizes, int n_in,
                              void* d_out, int out_size, void* d_ws, size_t ws_size,
                              hipStream_t stream) {
  (void)in_sizes; (void)n_in; (void)out_size; (void)ws_size;
  const float* x     = (const float*)d_in[0];
  const float* qw    = (const float*)d_in[1];
  const float* qb    = (const float*)d_in[2];
  const float* kw    = (const float*)d_in[3];
  const float* kbv   = (const float*)d_in[4];
  const float* ffw   = (const float*)d_in[5];
  const float* ffb   = (const float*)d_in[6];
  const float* gamma = (const float*)d_in[7];
  const float* beta  = (const float*)d_in[8];
  float* out = (float*)d_out;

  // workspace carve-out (~2.8 MB total)
  char* w = (char*)d_ws;
  auto carve = [&](size_t bytes) -> char* {
    char* p = w;
    w += (bytes + 255) & ~(size_t)255;
    return p;
  };
  float* qpart = (float*)carve(sizeof(float) * BB * NCHQ * FF);
  float* kpart = (float*)carve(sizeof(float) * BB * NCHQ * FF);
  float* qv    = (float*)carve(sizeof(float) * BB * FF);
  float* kv    = (float*)carve(sizeof(float) * BB * FF);
  unsigned short* attT = (unsigned short*)carve(sizeof(unsigned short) * BB * FF * FF);
  unsigned short* ffwb = (unsigned short*)carve(sizeof(unsigned short) * FF * FF);
  float* statp = (float*)carve(sizeof(float) * BB * NCHL * FF * 2);
  float* nA1 = (float*)carve(sizeof(float) * BB * FF);
  float* nC1 = (float*)carve(sizeof(float) * BB * FF);
  float* nA2 = (float*)carve(sizeof(float) * BB * FF);
  float* nC2 = (float*)carve(sizeof(float) * BB * FF);

  prep_ffw<<<(FF * FF + 255) / 256, 256, 0, stream>>>(ffw, ffwb);
  wave_partial<<<dim3(NCHQ, BB), 128, 0, stream>>>(x, qw, kw, qpart, kpart);
  wave_combine<<<BB, 128, 0, stream>>>(qpart, kpart, qw, qb, kw, kbv, qv, kv);
  att_softmax<<<BB, 128, 0, stream>>>(qv, kv, attT);

  // stage 1: y = x + x@att  (writes d_out + stats partials)
  gemm_fused<0><<<dim3(NCHL, BB), 256, 0, stream>>>(x, attT, nullptr, nullptr, nullptr, out, statp);
  stats_combine<<<BB, 128, 0, stream>>>(statp, gamma, beta, nA1, nC1);

  // stage 2: z = yn + yn@W^T + ffb  (in-place on d_out + stats partials)
  gemm_fused<1><<<dim3(NCHL, BB), 256, 0, stream>>>(out, ffwb, nA1, nC1, ffb, out, statp);
  stats_combine<<<BB, 128, 0, stream>>>(statp, gamma, beta, nA2, nC2);

  // final in-place instance norm
  final_norm<<<((BB * LL * FF / 4) + 255) / 256, 256, 0, stream>>>(out, nA2, nC2);
}
